// PointTransformer_19610820674067
// MI455X (gfx1250) — matrix operand order repaired and checked by reading, not compiled, not run on hardware
//
#include <hip/hip_runtime.h>
#include <math.h>

// ---------------------------------------------------------------------------
// PointTransformer forward for MI455X (gfx1250, wave32, WMMA).
// Every linear/conv1x1 goes through a guard-free WMMA GEMM: K is zero-padded
// to a multiple of 32 in workspace (exact for A@W^T), M is a multiple of 16
// (64 for the MT=4 path), N a multiple of 16. Per K-step, ALL raw b128 loads
// are issued first (one clause), then cvt_pk_f16, then MT independent v_wmma
// ops back-to-back into separate accumulators. __launch_bounds__(32,1) frees
// the register budget so fragments stay in distinct registers.
// ---------------------------------------------------------------------------

#define BATCH 2
#define NPC   2048
#define KNN   20
#define DW    512

typedef __attribute__((ext_vector_type(16))) _Float16 v16h;
typedef __attribute__((ext_vector_type(8)))  float    v8f;

static __device__ __forceinline__ v16h cvt16(const float4 a0, const float4 a1,
                                             const float4 a2, const float4 a3)
{
    v16h r;
    r[0]  = (_Float16)a0.x; r[1]  = (_Float16)a0.y;
    r[2]  = (_Float16)a0.z; r[3]  = (_Float16)a0.w;
    r[4]  = (_Float16)a1.x; r[5]  = (_Float16)a1.y;
    r[6]  = (_Float16)a1.z; r[7]  = (_Float16)a1.w;
    r[8]  = (_Float16)a2.x; r[9]  = (_Float16)a2.y;
    r[10] = (_Float16)a2.z; r[11] = (_Float16)a2.w;
    r[12] = (_Float16)a3.x; r[13] = (_Float16)a3.y;
    r[14] = (_Float16)a3.z; r[15] = (_Float16)a3.w;
    return r;
}

// ---------------------------------------------------------------------------
// WMMA GEMM: C[M,N] = act(A[M,K] @ W[N,K]^T + bias[N] (+ resid[M,N]))
// One wave per (MT x 16) x 16 output tile; B fragment shared across MT
// independent accumulators. Requires Kd % 32 == 0, M % (16*MT) == 0,
// N % 16 == 0 (guaranteed by host-side padding / dispatch).
// Fragment layouts per CDNA5 ISA 7.12.2 (16-bit A 16x32, B 32x16, f32 C/D).
// ---------------------------------------------------------------------------
template <int MT>
__global__ __launch_bounds__(32, 1) void k_gemm_wmma(
    const float* __restrict__ A, const float* __restrict__ W,
    const float* __restrict__ bias, const float* __restrict__ resid,
    float* __restrict__ C, int M, int N, int Kd, int doRelu)
{
    const int lane = threadIdx.x & 31;
    const int n0 = blockIdx.x * 16;          // N fastest: W stays L2-hot
    const int m0 = blockIdx.y * (16 * MT);

    const int l15  = lane & 15;
    const int hi   = lane >> 4;
    const int khiA = hi * 8;                 // lanes 16-31: K+8 / K+24 halves
    const int kbB  = hi * 8;                 // lanes 16-31: K+8 / K+24 halves
    const int bcol = n0 + l15;

    const float* Wp = W + (size_t)bcol * Kd;
    const float* Ap[MT];
#pragma unroll
    for (int t = 0; t < MT; ++t)
        Ap[t] = A + (size_t)(m0 + t * 16 + l15) * Kd;

    const v8f zacc = {};
    v8f acc[MT];
#pragma unroll
    for (int t = 0; t < MT; ++t) acc[t] = zacc;

    for (int k0 = 0; k0 < Kd; k0 += 32) {
        // ---- issue ALL raw loads for this K-step first (one clause) ----
        float4 ar[MT][4];
        float4 br[4];
#pragma unroll
        for (int t = 0; t < MT; ++t) {
            ar[t][0] = *(const float4*)(Ap[t] + k0 + khiA);
            ar[t][1] = *(const float4*)(Ap[t] + k0 + khiA + 4);
            ar[t][2] = *(const float4*)(Ap[t] + k0 + 16 + khiA);
            ar[t][3] = *(const float4*)(Ap[t] + k0 + 16 + khiA + 4);
        }
        br[0] = *(const float4*)(Wp + k0 + kbB);
        br[1] = *(const float4*)(Wp + k0 + kbB + 4);
        br[2] = *(const float4*)(Wp + k0 + 16 + kbB);
        br[3] = *(const float4*)(Wp + k0 + 16 + kbB + 4);

        // ---- convert to f16 fragments ----
        const v16h bf = cvt16(br[0], br[1], br[2], br[3]);
        v16h afh[MT];
#pragma unroll
        for (int t = 0; t < MT; ++t)
            afh[t] = cvt16(ar[t][0], ar[t][1], ar[t][2], ar[t][3]);

        // ---- MT independent WMMAs back-to-back (pipelined) ----
#pragma unroll
        for (int t = 0; t < MT; ++t) {
            acc[t] = __builtin_amdgcn_wmma_f32_16x16x32_f16(
                false, afh[t], false, bf, (short)0, acc[t], false, false);
        }
    }

    const float bv = bias ? bias[bcol] : 0.0f;
#pragma unroll
    for (int t = 0; t < MT; ++t) {
        const int rbase = m0 + t * 16 + hi * 8;   // lanes 16-31 -> rows +8
#pragma unroll
        for (int r = 0; r < 8; ++r) {
            const int row = rbase + r;
            float v = acc[t][r] + bv;
            if (resid) v += resid[(size_t)row * N + bcol];
            if (doRelu) v = fmaxf(v, 0.0f);
            C[(size_t)row * N + bcol] = v;
        }
    }
}

// zero-padded row copy: out[r, 0:Kin] = in[r, :], out[r, Kin:Kp] = 0
__global__ void k_pad_copy(const float* __restrict__ in, float* __restrict__ out,
                           int rows, int Kin, int Kp)
{
    size_t t = (size_t)blockIdx.x * blockDim.x + threadIdx.x;
    size_t total = (size_t)rows * Kp;
    if (t >= total) return;
    int c = (int)(t % Kp);
    size_t r = t / Kp;
    out[t] = (c < Kin) ? in[r * Kin + c] : 0.0f;
}

// ---------------------------------------------------------------------------
// kNN: KNN smallest squared distances, ascending (argsort incl. tie-break).
// ---------------------------------------------------------------------------
__global__ void k_knn(const float* __restrict__ dst, const float* __restrict__ src,
                      int* __restrict__ knn, int Bn, int M, int Nsrc)
{
    int t = blockIdx.x * blockDim.x + threadIdx.x;
    if (t >= Bn * M) return;
    int b = t / M;
    const float* dp = dst + (size_t)t * 3;
    const float  dx = dp[0], dy = dp[1], dz = dp[2];
    const float* sp = src + (size_t)b * Nsrc * 3;

    float prevV = -1.0f; int prevI = -1;
    for (int j = 0; j < KNN; ++j) {
        float bestV = 3.4e38f; int bestI = 0;
        for (int i = 0; i < Nsrc; ++i) {
            float ax = sp[i * 3 + 0] - dx;
            float ay = sp[i * 3 + 1] - dy;
            float az = sp[i * 3 + 2] - dz;
            float d  = ax * ax + ay * ay + az * az;
            bool after = (d > prevV) || (d == prevV && i > prevI);
            if (after && d < bestV) { bestV = d; bestI = i; }
        }
        knn[(size_t)t * KNN + j] = bestI;
        prevV = bestV; prevI = bestI;
    }
}

// ---------------------------------------------------------------------------
// Farthest point sampling, one block per batch (matches jax.lax.scan ref).
// ---------------------------------------------------------------------------
__global__ __launch_bounds__(256) void k_fps(const float* __restrict__ xyz,
                                             int* __restrict__ cent,
                                             float* __restrict__ dist,
                                             int Nsrc, int nsamp)
{
    const int b = blockIdx.x, tid = threadIdx.x;
    const float* p = xyz + (size_t)b * Nsrc * 3;
    float* d = dist + (size_t)b * Nsrc;
    for (int i = tid; i < Nsrc; i += 256) d[i] = 1e10f;

    __shared__ float sv[256];
    __shared__ int   si[256];
    __shared__ int   curIdx;
    if (tid == 0) { curIdx = 0; cent[b * nsamp] = 0; }
    __syncthreads();

    for (int s = 1; s < nsamp; ++s) {
        int c = curIdx;
        float cx = p[c * 3], cy = p[c * 3 + 1], cz = p[c * 3 + 2];
        float bv = -1.0f; int bi = 0;
        for (int i = tid; i < Nsrc; i += 256) {
            float ax = p[i * 3] - cx, ay = p[i * 3 + 1] - cy, az = p[i * 3 + 2] - cz;
            float nd = fminf(d[i], ax * ax + ay * ay + az * az);
            d[i] = nd;
            if (nd > bv) { bv = nd; bi = i; }
        }
        sv[tid] = bv; si[tid] = bi;
        __syncthreads();
        for (int o = 128; o > 0; o >>= 1) {
            if (tid < o) {
                if (sv[tid + o] > sv[tid] ||
                    (sv[tid + o] == sv[tid] && si[tid + o] < si[tid])) {
                    sv[tid] = sv[tid + o]; si[tid] = si[tid + o];
                }
            }
            __syncthreads();
        }
        if (tid == 0) { curIdx = si[0]; cent[b * nsamp + s] = si[0]; }
        __syncthreads();
    }
}

__global__ void k_gather_xyz(const float* __restrict__ xyz, const int* __restrict__ cent,
                             float* __restrict__ out, int Bn, int M2, int Nsrc)
{
    int t = blockIdx.x * blockDim.x + threadIdx.x;
    if (t >= Bn * M2 * 3) return;
    int c = t % 3, bm = t / 3, m = bm % M2, b = bm / M2;
    out[t] = xyz[((size_t)b * Nsrc + cent[b * M2 + m]) * 3 + c];
}

// diff rows padded to 32 floats: [0:3] = xyz[b,m]-xyz[b,knn], [3:32] = 0
__global__ void k_knn_diff_pad(const float* __restrict__ xyz, const int* __restrict__ knn,
                               float* __restrict__ out, int Bn, int M)
{
    size_t t = (size_t)blockIdx.x * blockDim.x + threadIdx.x;
    size_t total = (size_t)Bn * M * KNN * 32;
    if (t >= total) return;
    int c = (int)(t % 32);
    size_t r = t / 32;
    int j = (int)(r % KNN);
    size_t bm = r / KNN;
    int m = (int)(bm % M), b = (int)(bm / M);
    float v = 0.0f;
    if (c < 3) {
        int id = knn[bm * KNN + j];
        v = xyz[((size_t)b * M + m) * 3 + c] - xyz[((size_t)b * M + id) * 3 + c];
    }
    out[t] = v;
}

// t = q[:, :, None] - gather(Kall, knn) + pos
__global__ void k_qkpos(const float* __restrict__ q, const float* __restrict__ Kall,
                        const float* __restrict__ pos, const int* __restrict__ knn,
                        float* __restrict__ tout, int Bn, int M, int C)
{
    size_t t = (size_t)blockIdx.x * blockDim.x + threadIdx.x;
    size_t total = (size_t)Bn * M * KNN * C;
    if (t >= total) return;
    int c = (int)(t % C);
    size_t r = t / C;
    int j = (int)(r % KNN);
    size_t bm = r / KNN;
    int b = (int)(bm / M);
    int id = knn[bm * KNN + j];
    tout[t] = q[bm * C + c] - Kall[((size_t)b * M + id) * C + c] + pos[t];
}

// softmax over neighbors (axis=-2) of a*scale, then sum_j a*(vals[knn]+pos)
__global__ void k_attn_reduce(const float* __restrict__ a, const float* __restrict__ Vall,
                              const float* __restrict__ pos, const int* __restrict__ knn,
                              float* __restrict__ res, int Bn, int M, int C, float scale)
{
    size_t t = (size_t)blockIdx.x * blockDim.x + threadIdx.x;
    size_t total = (size_t)Bn * M * C;
    if (t >= total) return;
    int c = (int)(t % C);
    size_t bm = t / C;
    int b = (int)(bm / M);
    const float* ap = a   + bm * KNN * (size_t)C + c;
    const float* pp = pos + bm * KNN * (size_t)C + c;
    const int*   kp = knn + bm * KNN;

    float ev[KNN];
    float mx = -3.4e38f;
#pragma unroll
    for (int j = 0; j < KNN; ++j) { float v = ap[(size_t)j * C] * scale; ev[j] = v; mx = fmaxf(mx, v); }
    float sum = 0.0f;
#pragma unroll
    for (int j = 0; j < KNN; ++j) { float e = __expf(ev[j] - mx); ev[j] = e; sum += e; }
    float inv = 1.0f / sum;
    float accv = 0.0f;
#pragma unroll
    for (int j = 0; j < KNN; ++j) {
        int id = kp[j];
        accv += ev[j] * inv * (Vall[((size_t)b * M + id) * C + c] + pp[(size_t)j * C]);
    }
    res[t] = accv;
}

// grouped rows padded to CtPad (mult of 32): [0:3]=dxyz, [3:Ct]=feats, rest 0
__global__ void k_group_concat(const float* __restrict__ xyz, const float* __restrict__ nxyz,
                               const float* __restrict__ feats, const int* __restrict__ knn,
                               float* __restrict__ out, int Bn, int M2, int Cf, int Nsrc,
                               int CtPad)
{
    const int Ct = Cf + 3;
    size_t t = (size_t)blockIdx.x * blockDim.x + threadIdx.x;
    size_t total = (size_t)Bn * M2 * KNN * CtPad;
    if (t >= total) return;
    int c = (int)(t % CtPad);
    size_t r = t / CtPad;
    int j = (int)(r % KNN);
    size_t bm = r / KNN;
    int b = (int)(bm / M2);
    float v = 0.0f;
    if (c < Ct) {
        int id = knn[bm * KNN + j];
        if (c < 3) v = xyz[((size_t)b * Nsrc + id) * 3 + c] - nxyz[bm * 3 + c];
        else       v = feats[((size_t)b * Nsrc + id) * Cf + (c - 3)];
    }
    out[t] = v;
}

// per-channel mean/var over all rows (population var, matches jnp)
__global__ __launch_bounds__(256) void k_bn_stats(const float* __restrict__ x,
                                                  float* __restrict__ mv, int rows, int C)
{
    int c = blockIdx.x, tid = threadIdx.x;
    float s = 0.0f, s2 = 0.0f;
    for (int r = tid; r < rows; r += 256) {
        float v = x[(size_t)r * C + c];
        s += v; s2 += v * v;
    }
    __shared__ float sh[256], sh2[256];
    sh[tid] = s; sh2[tid] = s2;
    __syncthreads();
    for (int o = 128; o > 0; o >>= 1) {
        if (tid < o) { sh[tid] += sh[tid + o]; sh2[tid] += sh2[tid + o]; }
        __syncthreads();
    }
    if (tid == 0) {
        float m = sh[0] / rows;
        mv[c] = m;
        mv[C + c] = sh2[0] / rows - m * m;
    }
}

__global__ void k_bn_relu(float* __restrict__ x, const float* __restrict__ mv,
                          const float* __restrict__ g, const float* __restrict__ bta,
                          int rows, int C)
{
    size_t t = (size_t)blockIdx.x * blockDim.x + threadIdx.x;
    size_t total = (size_t)rows * C;
    if (t >= total) return;
    int c = (int)(t % C);
    float y = (x[t] - mv[c]) * rsqrtf(mv[C + c] + 1e-5f) * g[c] + bta[c];
    x[t] = fmaxf(y, 0.0f);
}

__global__ void k_maxk(const float* __restrict__ x, float* __restrict__ out,
                       int Bn, int M2, int C)
{
    size_t t = (size_t)blockIdx.x * blockDim.x + threadIdx.x;
    size_t total = (size_t)Bn * M2 * C;
    if (t >= total) return;
    int c = (int)(t % C);
    size_t bm = t / C;
    const float* p = x + bm * KNN * (size_t)C + c;
    float mx = p[0];
#pragma unroll
    for (int j = 1; j < KNN; ++j) mx = fmaxf(mx, p[(size_t)j * C]);
    out[t] = mx;
}

__global__ void k_max_points(const float* __restrict__ f, float* __restrict__ out,
                             int Bn, int M, int C)
{
    int t = blockIdx.x * blockDim.x + threadIdx.x;
    if (t >= Bn * C) return;
    int c = t % C, b = t / C;
    const float* p = f + (size_t)b * M * C + c;
    float mx = p[0];
    for (int m = 1; m < M; ++m) mx = fmaxf(mx, p[(size_t)m * C]);
    out[t] = mx;
}

// ---------------------------------------------------------------------------
// Host-side orchestration
// ---------------------------------------------------------------------------
struct TFp { const float *d1w,*d1b,*d2w,*d2b,*fc1w,*fc1b,*fc2w,*fc2b,
                          *g1w,*g1b,*g2w,*g2b,*wk,*wq,*wv; };
struct TDp { const float *bn1g,*bn1b,*bn2g,*bn2b,*c1w,*c1b,*c2w,*c2b; };

struct WB {
    int   *knn, *cent;
    float *fpsd, *diff, *x1, *q, *Kall, *Vall, *res;
    float *bufP, *bufT, *bufA;
    float *fA, *fB, *h1, *xyzA, *xyzB, *grp, *cv1, *cv2, *mv;
    float *xpad, *wpadA, *wpadC, *fc1awp;
};

static inline dim3 g1d(size_t total) { return dim3((unsigned)((total + 255) / 256)); }

// Requires Kd % 32 == 0, M % 16 == 0, N % 16 == 0.
static void gemm(hipStream_t s, const float* A, const float* W, const float* bias,
                 const float* resid, float* C, size_t M, int N, int Kd, int relu)
{
    if (M % 64 == 0) {
        dim3 g((unsigned)(N / 16), (unsigned)(M / 64), 1);
        k_gemm_wmma<4><<<g, dim3(32), 0, s>>>(A, W, bias, resid, C, (int)M, N, Kd, relu);
    } else {
        dim3 g((unsigned)(N / 16), (unsigned)(M / 16), 1);
        k_gemm_wmma<1><<<g, dim3(32), 0, s>>>(A, W, bias, resid, C, (int)M, N, Kd, relu);
    }
}

static void run_tf(hipStream_t s, const TFp& p, const float* xyz, const float* fin,
                   float* fout, int Bn, int M, int F, const WB& w)
{
    size_t BM = (size_t)Bn * M;
    k_knn<<<g1d(BM), 256, 0, s>>>(xyz, xyz, w.knn, Bn, M, M);
    gemm(s, fin, p.fc1w, p.fc1b, nullptr, w.x1, BM, DW, F, 0);
    gemm(s, w.x1, p.wq, nullptr, nullptr, w.q,    BM, DW, DW, 0);
    gemm(s, w.x1, p.wk, nullptr, nullptr, w.Kall, BM, DW, DW, 0);
    gemm(s, w.x1, p.wv, nullptr, nullptr, w.Vall, BM, DW, DW, 0);
    // pos branch: K=3 zero-padded to 32 (A rows and d1 weights)
    k_knn_diff_pad<<<g1d(BM * KNN * 32), 256, 0, s>>>(xyz, w.knn, w.diff, Bn, M);
    k_pad_copy<<<g1d((size_t)DW * 32), 256, 0, s>>>(p.d1w, w.wpadA, DW, 3, 32);
    gemm(s, w.diff, w.wpadA, p.d1b, nullptr, w.bufT, BM * KNN, DW, 32, 1);   // relu(d1)
    gemm(s, w.bufT, p.d2w, p.d2b, nullptr, w.bufP, BM * KNN, DW, DW, 0);     // pos
    k_qkpos<<<g1d(BM * KNN * DW), 256, 0, s>>>(w.q, w.Kall, w.bufP, w.knn, w.bufT, Bn, M, DW);
    gemm(s, w.bufT, p.g1w, p.g1b, nullptr, w.bufA, BM * KNN, DW, DW, 1);     // relu(g1)
    gemm(s, w.bufA, p.g2w, p.g2b, nullptr, w.bufT, BM * KNN, DW, DW, 0);     // a (pre-softmax)
    float scale = 1.0f / sqrtf((float)DW);
    k_attn_reduce<<<g1d(BM * DW), 256, 0, s>>>(w.bufT, w.Vall, w.bufP, w.knn, w.res, Bn, M, DW, scale);
    gemm(s, w.res, p.fc2w, p.fc2b, fin, fout, BM, F, DW, 0);                 // + residual feats
}

static void run_td(hipStream_t s, const TDp& p, const float* xyz, const float* fin,
                   int Bn, int Nsrc, int Cf, int C12, int M2,
                   float* nxyz, float* fout, const WB& w)
{
    k_fps<<<dim3(Bn), 256, 0, s>>>(xyz, w.cent, w.fpsd, Nsrc, M2);
    k_gather_xyz<<<g1d((size_t)Bn * M2 * 3), 256, 0, s>>>(xyz, w.cent, nxyz, Bn, M2, Nsrc);
    k_knn<<<g1d((size_t)Bn * M2), 256, 0, s>>>(nxyz, xyz, w.knn, Bn, M2, Nsrc);
    const int Ct = Cf + 3;
    const int CtPad = (Ct + 31) & ~31;                   // K padded to mult of 32
    size_t rows = (size_t)Bn * M2 * KNN;
    k_group_concat<<<g1d(rows * CtPad), 256, 0, s>>>(xyz, nxyz, fin, w.knn, w.grp, Bn, M2, Cf, Nsrc, CtPad);
    k_pad_copy<<<g1d((size_t)C12 * CtPad), 256, 0, s>>>(p.c1w, w.wpadC, C12, Ct, CtPad);
    gemm(s, w.grp, w.wpadC, p.c1b, nullptr, w.cv1, rows, C12, CtPad, 0);
    k_bn_stats<<<dim3(C12), 256, 0, s>>>(w.cv1, w.mv, (int)rows, C12);
    k_bn_relu<<<g1d(rows * C12), 256, 0, s>>>(w.cv1, w.mv, p.bn1g, p.bn1b, (int)rows, C12);
    gemm(s, w.cv1, p.c2w, p.c2b, nullptr, w.cv2, rows, C12, C12, 0);
    k_bn_stats<<<dim3(C12), 256, 0, s>>>(w.cv2, w.mv, (int)rows, C12);
    k_bn_relu<<<g1d(rows * C12), 256, 0, s>>>(w.cv2, w.mv, p.bn2g, p.bn2b, (int)rows, C12);
    k_maxk<<<g1d((size_t)Bn * M2 * C12), 256, 0, s>>>(w.cv2, fout, Bn, M2, C12);
}

extern "C" void kernel_launch(void* const* d_in, const int* in_sizes, int n_in,
                              void* d_out, int out_size, void* d_ws, size_t ws_size,
                              hipStream_t stream)
{
    (void)in_sizes; (void)n_in; (void)out_size; (void)ws_size;

    // ---- inputs: 'x' first, then params as sorted pytree leaves ----
    int ii = 0;
    auto nxt = [&]() { return (const float*)d_in[ii++]; };
    const float* X = nxt();                               // (B,1,N,3) == (B*N,3)
    const float* fc1a_w = nxt(); const float* fc1a_b = nxt();
    const float* fc1b_w = nxt(); const float* fc1b_b = nxt();
    TDp td[4];
    for (int t = 0; t < 4; ++t) {                         // keys sorted: bn1,bn2,c1,c2
        td[t].bn1g = nxt(); td[t].bn1b = nxt();
        td[t].bn2g = nxt(); td[t].bn2b = nxt();
        td[t].c1w  = nxt(); td[t].c1b  = nxt();
        td[t].c2w  = nxt(); td[t].c2b  = nxt();
    }
    TFp tf[5];
    for (int t = 0; t < 5; ++t) {                         // keys sorted: d1,d2,fc1,fc2,g1,g2,wk,wq,wv
        tf[t].d1w  = nxt(); tf[t].d1b  = nxt();
        tf[t].d2w  = nxt(); tf[t].d2b  = nxt();
        tf[t].fc1w = nxt(); tf[t].fc1b = nxt();
        tf[t].fc2w = nxt(); tf[t].fc2b = nxt();
        tf[t].g1w  = nxt(); tf[t].g1b  = nxt();
        tf[t].g2w  = nxt(); tf[t].g2b  = nxt();
        tf[t].wk   = nxt(); tf[t].wq   = nxt(); tf[t].wv = nxt();
    }

    // ---- workspace carve (bump allocator) ----
    char* wp = (char*)d_ws;
    auto aF = [&](size_t n) { float* p = (float*)wp; wp += ((n * 4 + 255) / 256) * 256; return p; };
    auto aI = [&](size_t n) { int*   p = (int*)wp;   wp += ((n * 4 + 255) / 256) * 256; return p; };

    const size_t BN  = (size_t)BATCH * NPC;               // 4096 points
    const size_t BNK = BN * KNN;                          // 81920 neighbor rows

    WB w;
    w.knn    = aI(BNK);
    w.cent   = aI((size_t)BATCH * 512);
    w.fpsd   = aF(BN);
    w.diff   = aF(BNK * 32);                              // K=3 padded to 32
    w.x1     = aF(BN * DW);
    w.q      = aF(BN * DW);
    w.Kall   = aF(BN * DW);
    w.Vall   = aF(BN * DW);
    w.res    = aF(BN * DW);
    w.bufP   = aF(BNK * DW);
    w.bufT   = aF(BNK * DW);
    w.bufA   = aF(BNK * DW);
    w.fA     = aF(BN * DW);
    w.fB     = aF(BN * DW);
    w.h1     = aF(BN * 32);
    w.xyzA   = aF(BN * 3);
    w.xyzB   = aF(BN * 3);
    w.grp    = aF((size_t)BATCH * 512 * KNN * 288);       // max CtPad = 288
    w.cv1    = aF((size_t)BATCH * 512 * KNN * DW);
    w.cv2    = aF((size_t)BATCH * 512 * KNN * DW);
    w.mv     = aF(2 * DW);
    w.xpad   = aF(BN * 32);                               // x padded 3 -> 32
    w.wpadA  = aF((size_t)DW * 32);                       // d1w padded 3 -> 32
    w.wpadC  = aF((size_t)DW * 288);                      // c1w padded Ct -> CtPad
    w.fc1awp = aF(32 * 32);                               // fc1a_w padded 3 -> 32

    hipStream_t s = stream;

    // ---- stem: f = fc1b(relu(fc1a(x))) with K zero-padded to 32 ----
    k_pad_copy<<<g1d(BN * 32), 256, 0, s>>>(X, w.xpad, (int)BN, 3, 32);
    k_pad_copy<<<g1d((size_t)32 * 32), 256, 0, s>>>(fc1a_w, w.fc1awp, 32, 3, 32);
    gemm(s, w.xpad, w.fc1awp, fc1a_b, nullptr, w.h1, BN, 32, 32, 1);
    gemm(s, w.h1, fc1b_w, fc1b_b, nullptr, w.fA, BN, 32, 32, 0);

    // ---- stage 1: tf1 @ 2048 pts, dim 32 -> td1 -> 512 pts, dim 64 ----
    run_tf(s, tf[0], X, w.fA, w.fB, BATCH, NPC, 32, w);
    run_td(s, td[0], X, w.fB, BATCH, NPC, 32, 64, 512, w.xyzA, w.fA, w);

    // ---- stage 2 ----
    run_tf(s, tf[1], w.xyzA, w.fA, w.fB, BATCH, 512, 64, w);
    run_td(s, td[1], w.xyzA, w.fB, BATCH, 512, 64, 128, 128, w.xyzB, w.fA, w);

    // ---- stage 3 ----
    run_tf(s, tf[2], w.xyzB, w.fA, w.fB, BATCH, 128, 128, w);
    run_td(s, td[2], w.xyzB, w.fB, BATCH, 128, 128, 256, 32, w.xyzA, w.fA, w);

    // ---- stage 4 ----
    run_tf(s, tf[3], w.xyzA, w.fA, w.fB, BATCH, 32, 256, w);
    run_td(s, td[3], w.xyzA, w.fB, BATCH, 32, 256, 512, 8, w.xyzB, w.fA, w);

    // ---- stage 5 + global max pool ----
    run_tf(s, tf[4], w.xyzB, w.fA, w.fB, BATCH, 8, 512, w);
    k_max_points<<<g1d((size_t)BATCH * DW), 256, 0, s>>>(w.fB, (float*)d_out, BATCH, 8, DW);
}
